// Decoder_48026324304188
// MI455X (gfx1250) — matrix operand order repaired and checked by reading, not compiled, not run on hardware
//
#include <hip/hip_runtime.h>
#include <hip/hip_bf16.h>
#include <math.h>

// ---------------- Types for WMMA (gfx1250, wave32) ----------------
typedef __attribute__((ext_vector_type(16))) __bf16 v16bf;
typedef __attribute__((ext_vector_type(8)))  float  v8f;

union BfFrag { v16bf v; uint4 q[2]; };

// Sizes
#define Bz 32
#define Sz 64
#define Tz 64
#define Hz 512
#define Dz 512
#define Vz 32000
#define G4H 2048           // 4*H
#define K0 1536            // emb(512)+out(512)+h0(512)
#define K1 1024            // h0n(512)+h1(512)

// float -> bf16, round-to-nearest-even
__device__ inline __bf16 f2bf(float f) {
  unsigned u = __builtin_bit_cast(unsigned, f);
  unsigned r = (u + 0x7FFFu + ((u >> 16) & 1u)) >> 16;
  unsigned short s = (unsigned short)r;
  return __builtin_bit_cast(__bf16, s);
}

__device__ inline float sigf(float x) { return 1.0f / (1.0f + expf(-x)); }

// A fragment: 16x32 bf16 per ISA layout (two 16B chunks, 32B apart).
__device__ inline v16bf load_a_frag(const __bf16* __restrict__ base) {
  BfFrag f;
  f.q[0] = *(const uint4*)(base);        // K {0..7} (lane<16) / {8..15}
  f.q[1] = *(const uint4*)(base + 16);   // K {16..23} / {24..31}
  return f.v;
}
// B fragment: 32x16 bf16; lane col = lane&15; two 16B chunks, 32B apart (same K order as A).
__device__ inline v16bf load_b_frag(const __bf16* __restrict__ base) {
  BfFrag f;
  f.q[0] = *(const uint4*)(base);
  f.q[1] = *(const uint4*)(base + 16);
  return f.v;
}

__device__ inline v8f wmma_bf16(v16bf a, v16bf b, v8f c) {
  return __builtin_amdgcn_wmma_f32_16x16x32_bf16(false, a, false, b, (short)0, c, false, false);
}

// ---------------- Weight conversion: fp32 -> bf16, optional concat ----------------
__global__ void conv_concat_kernel(__bf16* __restrict__ dst,
                                   const float* __restrict__ s0, int k0,
                                   const float* __restrict__ s1, int k1,
                                   long total) {
  long i = (long)blockIdx.x * blockDim.x + threadIdx.x;
  if (i >= total) return;
  int K = k0 + k1;
  long n = i / K;
  int  k = (int)(i - n * (long)K);
  float v = (k < k0) ? s0[n * (long)k0 + k] : s1[n * (long)k1 + (k - k0)];
  dst[i] = f2bf(v);
}

// ---------------- State init ----------------
__global__ void init_state_kernel(const float* __restrict__ h0, const float* __restrict__ c0,
                                  const float* __restrict__ out0,
                                  float* __restrict__ hstate, float* __restrict__ cstate,
                                  float* __restrict__ outstate) {
  int i = blockIdx.x * 256 + threadIdx.x;
  if (i < 2 * Bz * Hz) { hstate[i] = h0[i]; cstate[i] = c0[i]; }
  else {
    int j = i - 2 * Bz * Hz;
    if (j < Bz * Hz) outstate[j] = out0[j];
  }
}

// ---------------- Per-step activation assembly (fp32 -> bf16 concat) ----------------
__global__ void prep_step_kernel(const int* __restrict__ tokens, int t,
                                 const float* __restrict__ embed,
                                 const float* __restrict__ outstate,
                                 const float* __restrict__ hstate,
                                 __bf16* __restrict__ A0, __bf16* __restrict__ A1) {
  int i = blockIdx.x * 256 + threadIdx.x;
  if (i < Bz * K0) {
    int b = i / K0, k = i - b * K0;
    float v;
    if (k < 512) {
      int tok = tokens[t * Bz + b];
      v = (tok == 0) ? 0.0f : embed[(size_t)tok * Dz + k];   // padding_idx = 0
    } else if (k < 1024) {
      v = outstate[b * Hz + (k - 512)];
    } else {
      v = hstate[b * Hz + (k - 1024)];                        // layer-0 h
    }
    A0[i] = f2bf(v);
  } else {
    int j = i - Bz * K0;
    if (j < Bz * Hz) {
      int b = j / Hz, k = j - b * Hz;
      A1[b * K1 + 512 + k] = f2bf(hstate[Bz * Hz + b * Hz + k]); // layer-1 h
    }
  }
}

// ---------------- LSTM gates GEMM (bf16 WMMA) + fused cell nonlinearity ----------------
// K compile-time; register double-buffered pipeline; partial unroll keeps pressure bounded.
// grid: 8 blocks x 256 threads = 64 waves = 2 m_tiles x 32 n_tiles (H cols / 16).
template <int K>
__global__ __launch_bounds__(256, 1)
void lstm_gates_kernel(const __bf16* __restrict__ A,
                       const __bf16* __restrict__ W,
                       const float* __restrict__ b_ih, const float* __restrict__ b_hh,
                       float* __restrict__ hstate, float* __restrict__ cstate,
                       __bf16* __restrict__ hbf, int hbf_ld) {
  int wave = (blockIdx.x * blockDim.x + threadIdx.x) >> 5;
  int lane = threadIdx.x & 31;
  int m_tile = wave & 1;
  int n_tile = wave >> 1;                 // 0..31
  int hi   = lane & 15;
  int ksel_a = (lane < 16) ? 0 : 8;
  int ksel_b = (lane < 16) ? 0 : 8;
  int arow = m_tile * 16 + hi;

  v8f z = {0.f,0.f,0.f,0.f,0.f,0.f,0.f,0.f};
  v8f acc0 = z, acc1 = z, acc2 = z, acc3 = z;

  const __bf16* ap  = A + (size_t)arow * K + ksel_a;
  const __bf16* wp0 = W + (size_t)(0 * Hz + n_tile * 16 + hi) * K + ksel_b;
  const __bf16* wp1 = W + (size_t)(1 * Hz + n_tile * 16 + hi) * K + ksel_b;
  const __bf16* wp2 = W + (size_t)(2 * Hz + n_tile * 16 + hi) * K + ksel_b;
  const __bf16* wp3 = W + (size_t)(3 * Hz + n_tile * 16 + hi) * K + ksel_b;

  // software pipeline: load one k-step ahead of consumption
  v16bf ac  = load_a_frag(ap);
  v16bf bc0 = load_b_frag(wp0);
  v16bf bc1 = load_b_frag(wp1);
  v16bf bc2 = load_b_frag(wp2);
  v16bf bc3 = load_b_frag(wp3);

  constexpr int NIT = K / 32;
#pragma unroll 8
  for (int it = 0; it < NIT - 1; ++it) {
    int kn = (it + 1) * 32;
    v16bf an  = load_a_frag(ap  + kn);
    v16bf bn0 = load_b_frag(wp0 + kn);
    v16bf bn1 = load_b_frag(wp1 + kn);
    acc0 = wmma_bf16(ac, bc0, acc0);
    acc1 = wmma_bf16(ac, bc1, acc1);
    v16bf bn2 = load_b_frag(wp2 + kn);
    v16bf bn3 = load_b_frag(wp3 + kn);
    acc2 = wmma_bf16(ac, bc2, acc2);
    acc3 = wmma_bf16(ac, bc3, acc3);
    ac = an; bc0 = bn0; bc1 = bn1; bc2 = bn2; bc3 = bn3;
  }
  acc0 = wmma_bf16(ac, bc0, acc0);
  acc1 = wmma_bf16(ac, bc1, acc1);
  acc2 = wmma_bf16(ac, bc2, acc2);
  acc3 = wmma_bf16(ac, bc3, acc3);

  int h = n_tile * 16 + hi;                           // hidden index 0..511
  float bi  = b_ih[h]          + b_hh[h];
  float bff = b_ih[Hz + h]     + b_hh[Hz + h];
  float bg  = b_ih[2 * Hz + h] + b_hh[2 * Hz + h];
  float bo  = b_ih[3 * Hz + h] + b_hh[3 * Hz + h];
  int mbase = m_tile * 16 + ((lane < 16) ? 0 : 8);

#pragma unroll
  for (int r = 0; r < 8; ++r) {
    int m = mbase + r;                                // batch index 0..31
    float iv = sigf(acc0[r] + bi);
    float fv = sigf(acc1[r] + bff);
    float gv = tanhf(acc2[r] + bg);
    float ov = sigf(acc3[r] + bo);
    float cp = cstate[m * Hz + h];
    float c2 = fv * cp + iv * gv;
    float h2 = ov * tanhf(c2);
    cstate[m * Hz + h] = c2;
    hstate[m * Hz + h] = h2;
    if (hbf) hbf[(size_t)m * hbf_ld + h] = f2bf(h2);
  }
}

// ---------------- Attention + output projection (small; VALU fp32) ----------------
__global__ void attention_kernel(const float* __restrict__ hstate,     // [2,B,H]; layer1 at +B*H
                                 const float* __restrict__ context,    // [S,B,H]
                                 const float* __restrict__ W_in,       // [H,H]
                                 const float* __restrict__ W_out,      // [H,2H]
                                 float* __restrict__ outstate,         // [B,H]
                                 __bf16* __restrict__ hist,            // [T*B,H] bf16
                                 int t,
                                 float* __restrict__ attnbuf) {        // [B,S]
  int b = blockIdx.x, tid = threadIdx.x;
  __shared__ float h1[Hz], q[Hz], sc[Sz], wc[Hz];
  const float* h1g = hstate + Bz * Hz + b * Hz;
  for (int i = tid; i < Hz; i += 256) h1[i] = h1g[i];
  __syncthreads();
  for (int h = tid; h < Hz; h += 256) {
    const float* wr = W_in + (size_t)h * Hz;
    float s = 0.f;
    for (int k = 0; k < Hz; ++k) s += wr[k] * h1[k];
    q[h] = s;
  }
  __syncthreads();
  if (tid < Sz) {
    const float* cr = context + (size_t)tid * Bz * Hz + (size_t)b * Hz;
    float s = 0.f;
    for (int k = 0; k < Hz; ++k) s += cr[k] * q[k];
    sc[tid] = s;
  }
  __syncthreads();
  if (tid == 0) {
    float m = sc[0];
    for (int s = 1; s < Sz; ++s) m = fmaxf(m, sc[s]);
    float sum = 0.f;
    for (int s = 0; s < Sz; ++s) { float e = expf(sc[s] - m); sc[s] = e; sum += e; }
    float inv = 1.f / sum;
    for (int s = 0; s < Sz; ++s) sc[s] *= inv;
  }
  __syncthreads();
  if (tid < Sz) attnbuf[b * Sz + tid] = sc[tid];
  for (int h = tid; h < Hz; h += 256) {
    float s = 0.f;
    for (int sx = 0; sx < Sz; ++sx) s += sc[sx] * context[(size_t)sx * Bz * Hz + (size_t)b * Hz + h];
    wc[h] = s;
  }
  __syncthreads();
  for (int h = tid; h < Hz; h += 256) {
    const float* wr = W_out + (size_t)h * (2 * Hz);
    float s = 0.f;
    for (int k = 0; k < Hz; ++k) s += wr[k] * wc[k];
    for (int k = 0; k < Hz; ++k) s += wr[Hz + k] * h1[k];
    float o = tanhf(s);
    outstate[b * Hz + h] = o;
    hist[((size_t)t * Bz + b) * Hz + h] = f2bf(o);
  }
}

// ---------------- Generator GEMM: [2048,512]bf16 x [32000,512]bf16^T -> logits fp32 ----------------
// grid (250, 32); block 256 = 8 waves; block tile 64M x 128N; wave tile 32M x 32N (4 accums).
// A staged in LDS pre-swizzled into WMMA fragment layout: chunk index
//   ((mt*16 + ktIdx)*2 + plane)*32 + lane   (each (mt,ktIdx,plane) = contiguous 512B)
// B double-buffered in registers; uniform 16-step loop with unconditional prefetch
// (W_gen workspace is padded so the final dead prefetch stays in-bounds).
__global__ __launch_bounds__(256, 1)
void gen_gemm_kernel(const __bf16* __restrict__ A,     // [2048,512]
                     const __bf16* __restrict__ Wg,    // [32000,512] (+pad)
                     const float* __restrict__ bg,     // [32000]
                     float* __restrict__ Y) {          // [2048,32000]
  __shared__ uint4 sAu[4096];                       // 64KB: 64 rows x 512 K bf16, swizzled
  int tid = threadIdx.x;
  int wave = tid >> 5, lane = tid & 31;
  int mBase = blockIdx.y * 64;
  int nBase = blockIdx.x * 128;

  // Cooperative, coalesced A load; scatter into swizzled LDS layout.
  {
    const uint4* src = (const uint4*)(A + (size_t)mBase * Hz);  // 4096 x 16B, linear
#pragma unroll
    for (int it = 0; it < 16; ++it) {
      int i = it * 256 + tid;                 // linear src chunk index
      int row = i >> 6;                       // 0..63
      int c   = i & 63;                       // chunk within row (8 bf16 each)
      int half  = c & 1;                      // 0: K+0, 1: K+8
      int plane = (c >> 1) & 1;               // 0: K{0..15}, 1: K{16..31}
      int ktIdx = c >> 2;                     // 0..15
      int mt   = row >> 4;                    // 0..3
      int rlow = row & 15;
      sAu[((mt * 16 + ktIdx) * 2 + plane) * 32 + rlow + 16 * half] = src[i];
    }
  }
  __syncthreads();

  int m_half = wave & 1;                      // rows mBase + m_half*32
  int n_pair = wave >> 1;                     // 0..3 -> cols nBase + n_pair*32
  int hi = lane & 15;
  int kb = (lane < 16) ? 0 : 8;
  int mt0 = m_half * 2, mt1 = mt0 + 1;

  int nW = nBase + n_pair * 32;
  const __bf16* bp0 = Wg + (size_t)(nW      + hi) * Hz + kb;
  const __bf16* bp1 = Wg + (size_t)(nW + 16 + hi) * Hz + kb;

  v8f z = {0.f,0.f,0.f,0.f,0.f,0.f,0.f,0.f};
  v8f a00 = z, a01 = z, a10 = z, a11 = z;

  v16bf bc0 = load_b_frag(bp0);
  v16bf bc1 = load_b_frag(bp1);

#pragma unroll 8
  for (int kt = 0; kt < 16; ++kt) {
    int kn = (kt + 1) * 32;                   // kt==15 prefetch hits pad; never consumed
    v16bf bn0 = load_b_frag(bp0 + kn);
    v16bf bn1 = load_b_frag(bp1 + kn);

    BfFrag fa0, fa1;
    int base0 = ((mt0 * 16 + kt) * 2) * 32 + lane;
    int base1 = ((mt1 * 16 + kt) * 2) * 32 + lane;
    fa0.q[0] = sAu[base0]; fa0.q[1] = sAu[base0 + 32];
    fa1.q[0] = sAu[base1]; fa1.q[1] = sAu[base1 + 32];

    a00 = wmma_bf16(fa0.v, bc0, a00);
    a01 = wmma_bf16(fa0.v, bc1, a01);
    a10 = wmma_bf16(fa1.v, bc0, a10);
    a11 = wmma_bf16(fa1.v, bc1, a11);
    bc0 = bn0; bc1 = bn1;
  }

  int rsel = (lane < 16) ? 0 : 8;
  int mG0 = mBase + m_half * 32 + rsel;       // m tile 0 row base
  int mG1 = mG0 + 16;                         // m tile 1 row base
  int n0 = nW + hi, n1 = n0 + 16;
  float g0 = bg[n0], g1 = bg[n1];
#pragma unroll
  for (int r = 0; r < 8; ++r) {
    size_t row0 = (size_t)(mG0 + r) * Vz;
    size_t row1 = (size_t)(mG1 + r) * Vz;
    Y[row0 + n0] = a00[r] + g0;
    Y[row0 + n1] = a01[r] + g1;
    Y[row1 + n0] = a10[r] + g0;
    Y[row1 + n1] = a11[r] + g1;
  }
}

// ---------------- In-place row-wise log-softmax over V=32000 (float4 passes) ----------------
__global__ void logsoftmax_kernel(float* __restrict__ Y) {
  __shared__ float red[256];
  float4* p4 = (float4*)(Y + (size_t)blockIdx.x * Vz);
  const int N4 = Vz / 4;                      // 8000
  int tid = threadIdx.x;

  float mx = -INFINITY;
  for (int i = tid; i < N4; i += 256) {
    float4 v = p4[i];
    mx = fmaxf(mx, fmaxf(fmaxf(v.x, v.y), fmaxf(v.z, v.w)));
  }
  red[tid] = mx; __syncthreads();
  for (int s = 128; s > 0; s >>= 1) {
    if (tid < s) red[tid] = fmaxf(red[tid], red[tid + s]);
    __syncthreads();
  }
  mx = red[0]; __syncthreads();

  float sum = 0.f;
  for (int i = tid; i < N4; i += 256) {
    float4 v = p4[i];
    sum += expf(v.x - mx) + expf(v.y - mx) + expf(v.z - mx) + expf(v.w - mx);
  }
  red[tid] = sum; __syncthreads();
  for (int s = 128; s > 0; s >>= 1) {
    if (tid < s) red[tid] += red[tid + s];
    __syncthreads();
  }
  float lse = mx + logf(red[0]);

  for (int i = tid; i < N4; i += 256) {
    float4 v = p4[i];
    v.x -= lse; v.y -= lse; v.z -= lse; v.w -= lse;
    p4[i] = v;
  }
}

// ---------------- Final copy: hT, cT, last attn ----------------
__global__ void final_copy_kernel(const float* __restrict__ hstate,
                                  const float* __restrict__ cstate,
                                  const float* __restrict__ attnbuf,
                                  float* __restrict__ out) {
  const size_t OFF_H = (size_t)Tz * Bz * Vz;           // 65,536,000
  const size_t OFF_C = OFF_H + 2 * Bz * Hz;
  const size_t OFF_A = OFF_C + 2 * Bz * Hz;
  int i = blockIdx.x * 256 + threadIdx.x;
  if (i < 2 * Bz * Hz) out[OFF_H + i] = hstate[i];
  else if (i < 4 * Bz * Hz) out[OFF_C + (i - 2 * Bz * Hz)] = cstate[i - 2 * Bz * Hz];
  else if (i < 4 * Bz * Hz + Bz * Sz) out[OFF_A + (i - 4 * Bz * Hz)] = attnbuf[i - 4 * Bz * Hz];
}

// ---------------- Host launcher ----------------
extern "C" void kernel_launch(void* const* d_in, const int* in_sizes, int n_in,
                              void* d_out, int out_size, void* d_ws, size_t ws_size,
                              hipStream_t stream) {
  const int*   tokens = (const int*)  d_in[0];
  const float* h0     = (const float*)d_in[1];
  const float* c0     = (const float*)d_in[2];
  const float* context= (const float*)d_in[3];
  const float* out0   = (const float*)d_in[4];
  const float* embed  = (const float*)d_in[5];
  const float* W_ih0  = (const float*)d_in[6];
  const float* W_hh0  = (const float*)d_in[7];
  const float* b_ih0  = (const float*)d_in[8];
  const float* b_hh0  = (const float*)d_in[9];
  const float* W_ih1  = (const float*)d_in[10];
  const float* W_hh1  = (const float*)d_in[11];
  const float* b_ih1  = (const float*)d_in[12];
  const float* b_hh1  = (const float*)d_in[13];
  const float* W_in   = (const float*)d_in[14];
  const float* W_out  = (const float*)d_in[15];
  const float* W_gen  = (const float*)d_in[16];
  const float* b_gen  = (const float*)d_in[17];
  float* Y = (float*)d_out;

  // Workspace layout (all 256B aligned)
  char* ws = (char*)d_ws;
  size_t off = 0;
  __bf16* WCAT0 = (__bf16*)(ws + off); off += (size_t)G4H * K0 * 2;
  __bf16* WCAT1 = (__bf16*)(ws + off); off += (size_t)G4H * K1 * 2;
  __bf16* WGEN  = (__bf16*)(ws + off); off += (size_t)Vz * Hz * 2 + 256;  // +pad for tail prefetch
  __bf16* A0    = (__bf16*)(ws + off); off += (size_t)Bz * K0 * 2;
  __bf16* A1    = (__bf16*)(ws + off); off += (size_t)Bz * K1 * 2;
  float*  HST   = (float*)(ws + off);  off += (size_t)2 * Bz * Hz * 4;
  float*  CST   = (float*)(ws + off);  off += (size_t)2 * Bz * Hz * 4;
  float*  OST   = (float*)(ws + off);  off += (size_t)Bz * Hz * 4;
  __bf16* HIST  = (__bf16*)(ws + off); off += (size_t)Tz * Bz * Hz * 2;
  float*  ATTN  = (float*)(ws + off);  off += (size_t)Bz * Sz * 4;

  // 1) Weight conversions (bf16; [W_ih | W_hh] concatenated per LSTM layer)
  {
    long n0 = (long)G4H * K0;
    conv_concat_kernel<<<(unsigned)((n0 + 255) / 256), 256, 0, stream>>>(WCAT0, W_ih0, 1024, W_hh0, 512, n0);
    long n1 = (long)G4H * K1;
    conv_concat_kernel<<<(unsigned)((n1 + 255) / 256), 256, 0, stream>>>(WCAT1, W_ih1, 512, W_hh1, 512, n1);
    long ng = (long)Vz * Hz;
    conv_concat_kernel<<<(unsigned)((ng + 255) / 256), 256, 0, stream>>>(WGEN, W_gen, 512, (const float*)nullptr, 0, ng);
  }

  // 2) Init recurrent state
  init_state_kernel<<<(3 * Bz * Hz + 255) / 256, 256, 0, stream>>>(h0, c0, out0, HST, CST, OST);

  // 3) Sequential recurrence: T=64 steps
  for (int t = 0; t < Tz; ++t) {
    prep_step_kernel<<<(Bz * K0 + Bz * Hz + 255) / 256, 256, 0, stream>>>(tokens, t, embed, OST, HST, A0, A1);
    lstm_gates_kernel<K0><<<8, 256, 0, stream>>>(A0, WCAT0, b_ih0, b_hh0, HST, CST, A1, K1);
    lstm_gates_kernel<K1><<<8, 256, 0, stream>>>(A1, WCAT1, b_ih1, b_hh1,
                                                 HST + Bz * Hz, CST + Bz * Hz, (__bf16*)nullptr, 0);
    attention_kernel<<<Bz, 256, 0, stream>>>(HST, context, W_in, W_out, OST, HIST, t, ATTN);
  }

  // 4) Generator GEMM: [2048,512] x [512,32000] -> logits in d_out
  gen_gemm_kernel<<<dim3(Vz / 128, (Tz * Bz) / 64), 256, 0, stream>>>(HIST, WGEN, b_gen, Y);

  // 5) log-softmax in place, one block per row
  logsoftmax_kernel<<<Tz * Bz, 256, 0, stream>>>(Y);

  // 6) hT, cT, last attention
  final_copy_kernel<<<(4 * Bz * Hz + Bz * Sz + 255) / 256, 256, 0, stream>>>(HST, CST, ATTN, Y);
}
